// AutoResetRNN_70652212019733
// MI455X (gfx1250) — matrix operand order repaired and checked by reading, not compiled, not run on hardware
//
#include <hip/hip_runtime.h>
#include <hip/hip_bf16.h>
#include <cstdint>
#include <cstddef>

typedef __attribute__((ext_vector_type(16))) __bf16 v16bf;
typedef __attribute__((ext_vector_type(8)))  float  v8f;

static constexpr int T = 512, N = 256, D = 512, H = 512;
static constexpr int G3 = 3 * H;              // 1536 gate columns
static constexpr int NT = G3 / 16;            // 96 column tiles
static constexpr int KC = D / 32;             // 16 K-chunks of 32
static constexpr int PDW = 16;                // dwords per lane per chunk (8 hi + 8 lo)
static constexpr size_t PACK_DWORDS = (size_t)NT * KC * 32 * PDW; // 786432 (3 MB)
static constexpr int MROW = 128;              // gemm rows per WG (8 row-tiles)
static constexpr int RT = MROW / 16;          // 8 row tiles per WG
static constexpr int CB = 256;                // gemm cols per WG (16 col-tiles)

#define DEV static __device__ __forceinline__

// A-fragment K offset inside a 32-wide chunk (16-bit A 16x32 layout, ISA 7.12.2)
DEV int kA(int lane, int v) {
    return ((lane >> 4) & 1) * 8 + ((v >> 2) & 1) * 16 + (v & 3) * 2;
}
// B-fragment K offset (same K order as the A fragment)
DEV int kB(int lane, int v) {
    return ((lane >> 4) & 1) * 8 + ((v >> 2) & 1) * 16 + (v & 3) * 2;
}
DEV unsigned short bfbits(__bf16 h) { return __builtin_bit_cast(unsigned short, h); }

// Split two f32 values into packed bf16 hi-pair and lo-pair dwords.
DEV void split2(float a, float b, uint32_t& hi, uint32_t& lo) {
    __bf16 ah = (__bf16)a, bh = (__bf16)b;
    float ar = a - (float)ah, br = b - (float)bh;
    __bf16 al = (__bf16)ar, bl = (__bf16)br;
    hi = ((uint32_t)bfbits(bh) << 16) | (uint32_t)bfbits(ah);
    lo = ((uint32_t)bfbits(bl) << 16) | (uint32_t)bfbits(al);
}

union FragU { uint32_t d[8]; uint4 q[2]; v16bf v; };

DEV v8f wmma3(const FragU& ah, const FragU& al, const FragU& bh, const FragU& bl, v8f acc) {
    acc = __builtin_amdgcn_wmma_f32_16x16x32_bf16(false, ah.v, false, bh.v, (short)0, acc, false, false);
    acc = __builtin_amdgcn_wmma_f32_16x16x32_bf16(false, al.v, false, bh.v, (short)0, acc, false, false);
    acc = __builtin_amdgcn_wmma_f32_16x16x32_bf16(false, ah.v, false, bl.v, (short)0, acc, false, false);
    return acc;
}

// ---------------------------------------------------------------------------
// Kernel 0: repack w_ih / w_hh into B-fragment-major bf16 hi/lo buffers.
// ---------------------------------------------------------------------------
__global__ void pack_w(const float* __restrict__ w_ih, const float* __restrict__ w_hh,
                       uint32_t* __restrict__ pih, uint32_t* __restrict__ phh) {
    int tid = blockIdx.x * blockDim.x + threadIdx.x;
    const int per = NT * KC * 32;
    if (tid >= 2 * per) return;
    const float* W = (tid < per) ? w_ih : w_hh;
    uint32_t*    P = (tid < per) ? pih  : phh;
    int s = (tid < per) ? tid : tid - per;
    int lane = s & 31; s >>= 5;
    int kc = s % KC;   int j = s / KC;
    int n = lane & 15;
    const float* row = W + (size_t)(j * 16 + n) * D;   // B[k][n] = W[j*16+n][k]
    uint32_t* out = P + ((size_t)(j * KC + kc) * 32 + lane) * PDW;
#pragma unroll
    for (int v = 0; v < 8; ++v) {
        int k = kc * 32 + kB(lane, v);
        uint32_t hi, lo;
        split2(row[k], row[k + 1], hi, lo);
        out[v] = hi; out[8 + v] = lo;
    }
}

// ---------------------------------------------------------------------------
// Kernel 1: gx = x @ w_ih^T + b_ih   (131072 x 1536, K = 512)
// 2-D blocked: each WG computes 128 rows x 256 cols. 8 waves; each wave owns
// 2 column tiles x 8 row tiles (16 accumulators). A-fragments (hi/lo bf16 of
// the 128x32 x-chunk) are staged in LDS once per K-chunk and shared by all
// waves; each B fragment is reused across 8 row tiles.
// ---------------------------------------------------------------------------
__global__ void __launch_bounds__(256) gemm_gx(const float* __restrict__ x,
                                               const float* __restrict__ b_ih,
                                               const uint32_t* __restrict__ pih,
                                               float* __restrict__ gx) {
    __shared__ __align__(16) uint32_t afr[RT][32][16];   // 16 KB: hi d0-7, lo d8-15
    const int row0 = blockIdx.x * MROW;
    const int cb   = blockIdx.y;                         // 256-col block: 0..5
    const int lane = threadIdx.x & 31, wave = threadIdx.x >> 5;
    const int jt0  = cb * (CB / 16) + 2 * wave;          // this wave's first col tile

    v8f acc[RT][2];
#pragma unroll
    for (int rt = 0; rt < RT; ++rt) { acc[rt][0] = {}; acc[rt][1] = {}; }

    for (int kc = 0; kc < KC; ++kc) {
        // Stage A fragments of the 128x32 x-chunk (hi/lo bf16 split).
        for (int s = threadIdx.x; s < RT * 32 * 8; s += 256) {
            int v = s & 7, l = (s >> 3) & 31, rt = s >> 8;
            int m = l & 15;
            const float* xp = x + (size_t)(row0 + rt * 16 + m) * D + kc * 32 + kA(l, v);
            split2(xp[0], xp[1], afr[rt][l][v], afr[rt][l][8 + v]);
        }
        __syncthreads();

        FragU bh[2], bl[2];
#pragma unroll
        for (int c = 0; c < 2; ++c) {
            const uint4* bp = (const uint4*)(pih + ((size_t)((jt0 + c) * KC + kc) * 32 + lane) * PDW);
            bh[c].q[0] = bp[0]; bh[c].q[1] = bp[1];
            bl[c].q[0] = bp[2]; bl[c].q[1] = bp[3];
        }
#pragma unroll
        for (int rt = 0; rt < RT; ++rt) {
            FragU ah, al;
            const uint4* ap = (const uint4*)&afr[rt][lane][0];
            ah.q[0] = ap[0]; ah.q[1] = ap[1];
            al.q[0] = ap[2]; al.q[1] = ap[3];
            acc[rt][0] = wmma3(ah, al, bh[0], bl[0], acc[rt][0]);
            acc[rt][1] = wmma3(ah, al, bh[1], bl[1], acc[rt][1]);
        }
        __syncthreads();
    }

#pragma unroll
    for (int c = 0; c < 2; ++c) {
        int col = (jt0 + c) * 16 + (lane & 15);
        float bias = b_ih[col];
#pragma unroll
        for (int rt = 0; rt < RT; ++rt) {
#pragma unroll
            for (int r = 0; r < 8; ++r) {
                int m = r + 8 * (lane >> 4);
                gx[(size_t)(row0 + rt * 16 + m) * G3 + col] = acc[rt][c][r] + bias;
            }
        }
    }
}

// ---------------------------------------------------------------------------
// Kernel 2: GRU scan. 16 independent WGs, each owns 16 batch rows for all T.
// ---------------------------------------------------------------------------
__global__ void __launch_bounds__(512) gru_scan(const float* __restrict__ hxs,
                                                const float* __restrict__ masks,
                                                const float* __restrict__ b_hh,
                                                const uint32_t* __restrict__ phh,
                                                const float* __restrict__ gx,
                                                float* __restrict__ out) {
    __shared__ __align__(16) float    hA[16][H];         // 32 KB: hidden state (f32)
    __shared__ __align__(16) uint32_t aHi[KC][32][8];    // 16 KB: A-frag hi
    __shared__ __align__(16) uint32_t aLo[KC][32][8];    // 16 KB: A-frag lo

    const int row0 = blockIdx.x * 16;
    const int tid = threadIdx.x;
    const int lane = tid & 31, wave = tid >> 5;

    for (int i = tid; i < 16 * H; i += 512) {
        int m = i >> 9, c = i & (H - 1);
        hA[m][c] = hxs[(size_t)(row0 + m) * H + c];
    }
    __syncthreads();

    for (int t = 0; t < T; ++t) {
        const int rbase = t * N + row0;

        // Phase A: apply mask and build bf16 hi/lo A fragments of h (16x512).
        for (int s = tid; s < KC * 32 * 8; s += 512) {
            int v = s & 7, l = (s >> 3) & 31, kc = s >> 8;
            int m = l & 15, k = kc * 32 + kA(l, v);
            float mk = masks[rbase + m];
            split2(hA[m][k] * mk, hA[m][k + 1] * mk, aHi[kc][l][v], aLo[kc][l][v]);
        }
        __syncthreads();

        // Phase B: gh = h @ w_hh^T for this wave's column tiles + fused gates.
#pragma unroll
        for (int ci = 0; ci < 2; ++ci) {
            const int ct = wave + 16 * ci;               // column tile 0..31
            v8f aR = {}, aZ = {}, aN = {};
            for (int kc = 0; kc < KC; ++kc) {
                FragU ah, al;
                const uint4* ap  = (const uint4*)&aHi[kc][lane][0];
                const uint4* alp = (const uint4*)&aLo[kc][lane][0];
                ah.q[0] = ap[0];  ah.q[1] = ap[1];
                al.q[0] = alp[0]; al.q[1] = alp[1];
#pragma unroll
                for (int g = 0; g < 3; ++g) {
                    int j = ct + 32 * g;                 // r / z / n gate tile
                    const uint4* bp = (const uint4*)(phh + ((size_t)(j * KC + kc) * 32 + lane) * PDW);
                    FragU bh, bl;
                    bh.q[0] = bp[0]; bh.q[1] = bp[1];
                    bl.q[0] = bp[2]; bl.q[1] = bp[3];
                    if (g == 0)      aR = wmma3(ah, al, bh, bl, aR);
                    else if (g == 1) aZ = wmma3(ah, al, bh, bl, aZ);
                    else             aN = wmma3(ah, al, bh, bl, aN);
                }
            }
            const int col = ct * 16 + (lane & 15);
            const float bR = b_hh[col], bZ = b_hh[col + H], bN = b_hh[col + 2 * H];
#pragma unroll
            for (int r = 0; r < 8; ++r) {
                int m = r + 8 * (lane >> 4);
                int grow = rbase + m;
                size_t gb = (size_t)grow * G3;
                float xr = gx[gb + col], xz = gx[gb + H + col], xn = gx[gb + 2 * H + col];
                float mk = masks[grow];
                float hold = hA[m][col] * mk;
                float rr = 1.f / (1.f + __expf(-(xr + aR[r] + bR)));
                float zz = 1.f / (1.f + __expf(-(xz + aZ[r] + bZ)));
                float nn = tanhf(xn + rr * (aN[r] + bN));
                float hnew = (1.f - zz) * nn + zz * hold;
                hA[m][col] = hnew;                        // only this lane touches (m,col)
                out[(size_t)grow * H + col] = hnew;       // ys[t,n,:]
            }
        }
        __syncthreads();
    }

    const size_t TNH = (size_t)T * N * H;
    for (int i = tid; i < 16 * H; i += 512) {
        int m = i >> 9, c = i & (H - 1);
        out[TNH + (size_t)(row0 + m) * H + c] = hA[m][c];  // h_last
    }
}

// ---------------------------------------------------------------------------
extern "C" void kernel_launch(void* const* d_in, const int* in_sizes, int n_in,
                              void* d_out, int out_size, void* d_ws, size_t ws_size,
                              hipStream_t stream) {
    (void)in_sizes; (void)n_in; (void)out_size; (void)ws_size;
    const float* x     = (const float*)d_in[0];
    const float* hxs   = (const float*)d_in[1];
    const float* masks = (const float*)d_in[2];
    const float* w_ih  = (const float*)d_in[3];
    const float* w_hh  = (const float*)d_in[4];
    const float* b_ih  = (const float*)d_in[5];
    const float* b_hh  = (const float*)d_in[6];

    uint32_t* pih = (uint32_t*)d_ws;                 // 3 MB packed w_ih
    uint32_t* phh = pih + PACK_DWORDS;               // 3 MB packed w_hh
    float*    gx  = (float*)(phh + PACK_DWORDS);     // 768 MB gx = x@w_ih^T + b_ih
    float*    out = (float*)d_out;

    const int packThreads = 2 * NT * KC * 32;        // 98304
    pack_w<<<(packThreads + 255) / 256, 256, 0, stream>>>(w_ih, w_hh, pih, phh);
    dim3 ggrid((T * N) / MROW, G3 / CB);             // 1024 x 6
    gemm_gx<<<ggrid, 256, 0, stream>>>(x, b_ih, pih, gx);
    gru_scan<<<N / 16, 512, 0, stream>>>(hxs, masks, b_hh, phh, gx, out);
}
